// Model_52725018525789
// MI455X (gfx1250) — matrix operand order repaired and checked by reading, not compiled, not run on hardware
//
#include <hip/hip_runtime.h>
#include <stdint.h>

#define BATCH   4
#define EPSV    1e-5f
#define NSWEEP  6

typedef __attribute__((ext_vector_type(16))) __bf16 v16bf;
typedef __attribute__((ext_vector_type(8)))  float  v8f;
typedef __attribute__((ext_vector_type(2)))  float  v2f;
typedef __attribute__((ext_vector_type(4)))  unsigned int u32x4;
typedef __attribute__((ext_vector_type(8)))  int    i32x8;
typedef __attribute__((ext_vector_type(4)))  int    i32x4;

// ---------------------------------------------------------------------------
// TDM: async 2-D f32 tile load, global -> LDS (D# packed per CDNA5 ISA 8.3/8.4)
// ---------------------------------------------------------------------------
__device__ __forceinline__ void tdm_load_2d_f32(
    uint32_t lds_off, const float* gsrc,
    uint32_t tdim0, uint32_t tdim1, uint32_t tile0, uint32_t tile1,
    uint64_t stride0_elems)
{
  uint64_t ga = (uint64_t)(uintptr_t)gsrc;
  u32x4 g0;
  g0[0] = 1u;                                   // count=1, user descriptor
  g0[1] = lds_off;                              // LDS byte address
  g0[2] = (uint32_t)ga;                         // global_addr[31:0]
  g0[3] = (uint32_t)((ga >> 32) & 0x01ffffffu)  // global_addr[56:32]
        | (2u << 30);                           // type = 2 ("image")
  i32x8 g1;
  g1[0] = (int)(2u << 16);                      // data_size = 4 bytes
  g1[1] = (int)((tdim0 & 0xffffu) << 16);       // tensor_dim0[15:0] @ bits63:48
  g1[2] = (int)((tdim0 >> 16) | ((tdim1 & 0xffffu) << 16));
  g1[3] = (int)((tdim1 >> 16) | (tile0 << 16)); // tile_dim0
  g1[4] = (int)(tile1 & 0xffffu);               // tile_dim1 (tile_dim2 = 0)
  g1[5] = (int)(uint32_t)(stride0_elems & 0xffffffffu);
  g1[6] = (int)(uint32_t)((stride0_elems >> 32) & 0xffffu);
  g1[7] = 0;
  i32x4 zz; zz[0]=0; zz[1]=0; zz[2]=0; zz[3]=0;
#if __clang_major__ >= 23
  i32x8 z8; z8[0]=0; z8[1]=0; z8[2]=0; z8[3]=0; z8[4]=0; z8[5]=0; z8[6]=0; z8[7]=0;
  __builtin_amdgcn_tensor_load_to_lds(g0, g1, zz, zz, z8, 0);
#else
  __builtin_amdgcn_tensor_load_to_lds(g0, g1, zz, zz, 0);
#endif
}

// ---------------------------------------------------------------------------
// 3x3 conv, reflect pad, implicit GEMM: M=Cout, N=H*W, K=Cin*9.
// bf16 WMMA 16x16x32, f32 accumulate. A-tile (weights) via TDM into LDS.
// Block = 256 threads (8 waves): wave computes one 16(M) x 32(N) strip.
// ---------------------------------------------------------------------------
__global__ __launch_bounds__(256) void k_conv3x3(
    const float* __restrict__ in, const float* __restrict__ wgt,
    const float* __restrict__ bias, float* __restrict__ out,
    int Cin, int Cout, int H, int W, int do_relu)
{
  extern __shared__ char smem[];
  float*  As = (float*)smem;                 // [64][32] f32 weight tile (TDM dest, LDS off 0)
  __bf16* Bs = (__bf16*)(smem + 64*32*4);    // [64 pix][32 k] bf16 activation tile

  const int HW = H * W;
  const int K  = Cin * 9;
  const int b  = blockIdx.z;
  const int m0 = blockIdx.y * 64;
  const int n0 = blockIdx.x * 64;
  const int tid  = threadIdx.x;
  const int lane = tid & 31;
  const int wv   = tid >> 5;
  const int mt   = wv & 3;
  const int nh   = wv >> 2;
  const int l15  = lane & 15;

  v8f acc0 = {0.f,0.f,0.f,0.f,0.f,0.f,0.f,0.f};
  v8f acc1 = acc0;

  const int nK = (K + 31) >> 5;
  for (int kb = 0; kb < nK; ++kb) {
    const int k0 = kb * 32;
    __syncthreads();                        // LDS safe to overwrite
    if (wv == 0) {                          // one wave drives the TDM
      tdm_load_2d_f32(0u, wgt + (size_t)m0 * K + k0,
                      (uint32_t)(K - k0), (uint32_t)(Cout - m0),
                      32u, 64u, (uint64_t)K);
      __builtin_amdgcn_s_wait_tensorcnt(0);
    }
    // Activation gather (implicit-GEMM + reflect pad), f32 -> bf16 into LDS.
    for (int i = tid; i < 64 * 32; i += 256) {
      int tn = i >> 5, tk = i & 31;
      int k = k0 + tk;
      float v = 0.f;
      if (k < K) {
        int ci = k / 9, r = (k % 9) / 3, c = k % 3;
        int p  = n0 + tn;
        int py = p / W;
        int y  = py + r - 1;
        int x  = (p - py * W) + c - 1;
        y = (y < 0) ? -y : ((y >= H) ? 2 * H - 2 - y : y);
        x = (x < 0) ? -x : ((x >= W) ? 2 * W - 2 - x : x);
        v = in[((size_t)b * Cin + ci) * HW + (size_t)y * W + x];
      }
      Bs[tn * 32 + tk] = (__bf16)v;
    }
    __syncthreads();
    // Fragments: A and B both use contiguous k order (k = 16*(lane/16) + s).
    const int row = mt * 16 + l15;
    const int ka  = (lane < 16) ? 0 : 16;
    v16bf af;
#pragma unroll
    for (int t = 0; t < 8; ++t) af[t]     = (__bf16)As[row * 32 + ka + t];
#pragma unroll
    for (int t = 0; t < 8; ++t) af[t + 8] = (__bf16)As[row * 32 + ka + 8 + t];
    const int kbb = (lane < 16) ? 0 : 16;
    const v16bf bf0 = *(const v16bf*)(Bs + ((nh * 2 + 0) * 16 + l15) * 32 + kbb);
    const v16bf bf1 = *(const v16bf*)(Bs + ((nh * 2 + 1) * 16 + l15) * 32 + kbb);
    acc0 = __builtin_amdgcn_wmma_f32_16x16x32_bf16(false, af, false, bf0, (short)0, acc0, false, false);
    acc1 = __builtin_amdgcn_wmma_f32_16x16x32_bf16(false, af, false, bf1, (short)0, acc1, false, false);
  }
  // Epilogue: bias + optional ReLU. C/D layout: VGPR r -> M=16*mt+r (+8 for lanes>=16).
  const int mbase = m0 + mt * 16 + ((lane < 16) ? 0 : 8);
  const int nbase = n0 + nh * 32 + l15;
#pragma unroll
  for (int r = 0; r < 8; ++r) {
    int m = mbase + r;
    if (m < Cout) {
      float bv = bias[m];
      float v0 = acc0[r] + bv, v1 = acc1[r] + bv;
      if (do_relu) { v0 = fmaxf(v0, 0.f); v1 = fmaxf(v1, 0.f); }
      size_t base = ((size_t)b * Cout + m) * HW;
      out[base + nbase]      = v0;
      out[base + nbase + 16] = v1;
    }
  }
}

// ---------------------------------------------------------------------------
// f32 GEMM via V_WMMA_F32_16X16X4_F32 (full precision for WCT / eigensolve).
// C[M,N] = scale * A[M,K] @ B ; transB: B given row-major [N,K] (i.e. use B^T).
// Optional runtime scale (1/(n-1)) read from scale_ptr; diag_eps added on diag.
// ---------------------------------------------------------------------------
__global__ __launch_bounds__(256) void k_gemm_f32(
    const float* __restrict__ A, const float* __restrict__ B, float* __restrict__ C,
    int M, int N, int Kd, int transB,
    const float* __restrict__ scale_ptr, float scale, float diag_eps)
{
  __shared__ float As[64 * 32];
  __shared__ float Bt[64 * 32];           // Bt[n][k]
  const int m0 = blockIdx.y * 64;
  const int n0 = blockIdx.x * 64;
  const int tid  = threadIdx.x;
  const int lane = tid & 31;
  const int wv   = tid >> 5;
  const int mt   = wv & 3;
  const int nh   = wv >> 2;
  const int l15  = lane & 15;
  const int row  = mt * 16 + l15;

  v8f acc0 = {0.f,0.f,0.f,0.f,0.f,0.f,0.f,0.f};
  v8f acc1 = acc0;

  for (int k0 = 0; k0 < Kd; k0 += 32) {
    __syncthreads();
    for (int i = tid; i < 64 * 32; i += 256) {
      int rr = i >> 5, kk = i & 31;
      int gm = m0 + rr, gk = k0 + kk, gn = n0 + rr;
      As[i] = (gm < M && gk < Kd) ? A[(size_t)gm * Kd + gk] : 0.f;
      float bv = 0.f;
      if (gn < N && gk < Kd)
        bv = transB ? B[(size_t)gn * Kd + gk] : B[(size_t)gk * N + gn];
      Bt[i] = bv;
    }
    __syncthreads();
#pragma unroll
    for (int kk = 0; kk < 8; ++kk) {
      const int ks = kk * 4 + ((lane < 16) ? 0 : 2);  // f32 A 16x4 / B 4x16 layout
      v2f a, b0, b1;
      a[0]  = As[row * 32 + ks];  a[1]  = As[row * 32 + ks + 1];
      const int nA = nh * 32 + l15, nB = nA + 16;
      b0[0] = Bt[nA * 32 + ks];   b0[1] = Bt[nA * 32 + ks + 1];
      b1[0] = Bt[nB * 32 + ks];   b1[1] = Bt[nB * 32 + ks + 1];
      acc0 = __builtin_amdgcn_wmma_f32_16x16x4_f32(false, a, false, b0, (short)0, acc0, false, false);
      acc1 = __builtin_amdgcn_wmma_f32_16x16x4_f32(false, a, false, b1, (short)0, acc1, false, false);
    }
  }
  const float sc = scale_ptr ? scale_ptr[0] : scale;
  const int mbase = m0 + mt * 16 + ((lane < 16) ? 0 : 8);
  const int nb0   = n0 + nh * 32 + l15;
#pragma unroll
  for (int r = 0; r < 8; ++r) {
    int gm = mbase + r;
    if (gm < M) {
      if (nb0 < N)
        C[(size_t)gm * N + nb0] = acc0[r] * sc + ((diag_eps > 0.f && gm == nb0) ? diag_eps : 0.f);
      if (nb0 + 16 < N)
        C[(size_t)gm * N + nb0 + 16] = acc1[r] * sc + ((diag_eps > 0.f && gm == nb0 + 16) ? diag_eps : 0.f);
    }
  }
}

// ---------------------------------------------------------------------------
// Parallel cyclic Jacobi: one sweep (n-1 rounds, n/2 disjoint pairs per round).
// One workgroup, n=512, 256 threads = 256 pairs. A' = J^T A J ; V' = V J.
// ---------------------------------------------------------------------------
__global__ __launch_bounds__(256) void k_jacobi_sweep(float* A, float* V, int n)
{
  const int t = threadIdx.x;
  for (int r = 0; r < n - 1; ++r) {
    int s0 = (t == 0) ? 0 : ((t - 1 + r) % (n - 1)) + 1;     // seat t
    int s1 = ((n - 2 - t + r) % (n - 1)) + 1;                // seat n-1-t
    int p = s0 < s1 ? s0 : s1;
    int q = s0 < s1 ? s1 : s0;
    float app = A[(size_t)p * n + p], aqq = A[(size_t)q * n + q], apq = A[(size_t)p * n + q];
    float c = 1.f, s = 0.f;
    if (fabsf(apq) > 1e-12f) {
      float tau = (aqq - app) / (2.f * apq);
      float tt  = ((tau >= 0.f) ? 1.f : -1.f) / (fabsf(tau) + sqrtf(1.f + tau * tau));
      c = rsqrtf(1.f + tt * tt);
      s = tt * c;
    }
    __syncthreads();
    for (int j = 0; j < n; ++j) {          // rows p,q : J^T A
      float ap = A[(size_t)p * n + j], aq = A[(size_t)q * n + j];
      A[(size_t)p * n + j] = c * ap - s * aq;
      A[(size_t)q * n + j] = s * ap + c * aq;
    }
    __syncthreads();
    for (int i = 0; i < n; ++i) {          // cols p,q : (.)J ; V = V J
      float ap = A[(size_t)i * n + p], aq = A[(size_t)i * n + q];
      A[(size_t)i * n + p] = c * ap - s * aq;
      A[(size_t)i * n + q] = s * ap + c * aq;
      float vp = V[(size_t)i * n + p], vq = V[(size_t)i * n + q];
      V[(size_t)i * n + p] = c * vp - s * vq;
      V[(size_t)i * n + q] = s * vp + c * vq;
    }
    __syncthreads();
  }
}

// ---------------------------------------------------------------------------
// small utility kernels
// ---------------------------------------------------------------------------
__global__ void k_zero(float* p, long long n) {
  long long i = (long long)blockIdx.x * blockDim.x + threadIdx.x;
  long long st = (long long)gridDim.x * blockDim.x;
  for (; i < n; i += st) p[i] = 0.f;
}
__global__ void k_identity(float* V, int n) {
  int i = blockIdx.x * blockDim.x + threadIdx.x;
  if (i < n * n) V[i] = ((i / n) == (i % n)) ? 1.f : 0.f;
}
__global__ void k_conv1x1(const float* __restrict__ in, const float* __restrict__ w,
                          const float* __restrict__ b, float* __restrict__ out, int HW) {
  int i = blockIdx.x * blockDim.x + threadIdx.x;
  int total = BATCH * 3 * HW;
  if (i >= total) return;
  int p = i % HW, co = (i / HW) % 3, bb = i / (HW * 3);
  float s = b[co];
  for (int ci = 0; ci < 3; ++ci) s += w[co * 3 + ci] * in[((size_t)bb * 3 + ci) * HW + p];
  out[i] = s;
}
__global__ void k_pool2(const float* __restrict__ in, float* __restrict__ out, int C, int H, int W) {
  int H2 = H >> 1, W2 = W >> 1;
  int i = blockIdx.x * blockDim.x + threadIdx.x;
  int total = BATCH * C * H2 * W2;
  if (i >= total) return;
  int x = i % W2, y = (i / W2) % H2, c = i / (W2 * H2);
  const float* p = in + ((size_t)c * H + 2 * y) * W + 2 * x;
  float m = fmaxf(fmaxf(p[0], p[1]), fmaxf(p[W], p[W + 1]));
  out[i] = m;
}
__global__ void k_up2(const float* __restrict__ in, float* __restrict__ out, int C, int H, int W) {
  int H2 = H << 1, W2 = W << 1;
  int i = blockIdx.x * blockDim.x + threadIdx.x;
  int total = BATCH * C * H2 * W2;
  if (i >= total) return;
  int x = i % W2, y = (i / W2) % H2, c = i / (W2 * H2);
  out[i] = in[((size_t)c * H + (y >> 1)) * W + (x >> 1)];
}
// masked per-channel mean + centered/masked features for one (batch, cluster)
__global__ __launch_bounds__(256) void k_masked_center(
    const float* __restrict__ f, const int* __restrict__ lab,
    float* __restrict__ X, float* __restrict__ meanv, float* __restrict__ invnm1,
    int b, int klus, int C, int Np)
{
  __shared__ float red[256];
  __shared__ float s_mean;
  int c = blockIdx.x, t = threadIdx.x;
  const float* fc = f + ((size_t)b * C + c) * Np;
  const int* lb = lab + (size_t)b * Np;
  float sum = 0.f, cnt = 0.f;
  for (int n = t; n < Np; n += 256) {
    float m = (lb[n] == klus) ? 1.f : 0.f;
    sum += fc[n] * m; cnt += m;
  }
  red[t] = sum; __syncthreads();
  for (int s = 128; s > 0; s >>= 1) { if (t < s) red[t] += red[t + s]; __syncthreads(); }
  float tot = red[0]; __syncthreads();
  red[t] = cnt; __syncthreads();
  for (int s = 128; s > 0; s >>= 1) { if (t < s) red[t] += red[t + s]; __syncthreads(); }
  if (t == 0) {
    float nn = red[0]; if (nn < 2.f) nn = 2.f;
    s_mean = tot / nn;
    meanv[c] = s_mean;
    if (c == 0) invnm1[0] = 1.f / (nn - 1.f);
  }
  __syncthreads();
  float mu = s_mean;
  for (int n = t; n < Np; n += 256) {
    float m = (lb[n] == klus) ? 1.f : 0.f;
    X[(size_t)c * Np + n] = (fc[n] - mu) * m;
  }
}
// VD = V * diag(clamp(diag(A),eps)^power) ; VT = V^T
__global__ void k_eig_post(const float* __restrict__ A, const float* __restrict__ V,
                           float* __restrict__ VD, float* __restrict__ VT, int n, float power)
{
  int i = blockIdx.x * blockDim.x + threadIdx.x;
  if (i >= n * n) return;
  int r = i / n, j = i % n;
  float d = A[(size_t)j * n + j];
  float f = __powf(fmaxf(d, EPSV), power);
  float v = V[(size_t)r * n + j];
  VD[(size_t)r * n + j] = v * f;
  VT[(size_t)j * n + r] = v;
}
// cs_b += colored + smean (x) cmask    (alpha == 1)
__global__ void k_wct_accum(float* __restrict__ cs, const float* __restrict__ colored,
                            const float* __restrict__ smean, const int* __restrict__ lab,
                            int b, int klus, int C, int Np)
{
  int i = blockIdx.x * blockDim.x + threadIdx.x;
  if (i >= C * Np) return;
  int c = i / Np, n = i % Np;
  float cm = (lab[(size_t)b * Np + n] == klus) ? 1.f : 0.f;
  cs[i] += colored[i] + smean[c] * cm;
}
// per-(b,c) mean & unbiased std over HW
__global__ __launch_bounds__(256) void k_meanstd(const float* __restrict__ f,
                                                 float* __restrict__ mean, float* __restrict__ sd,
                                                 int C, int HW)
{
  __shared__ float r0[256], r1[256];
  int bc = blockIdx.x, t = threadIdx.x;
  const float* p = f + (size_t)bc * HW;
  float s = 0.f, s2 = 0.f;
  for (int i = t; i < HW; i += 256) { float v = p[i]; s += v; s2 += v * v; }
  r0[t] = s; r1[t] = s2; __syncthreads();
  for (int st = 128; st > 0; st >>= 1) {
    if (t < st) { r0[t] += r0[t + st]; r1[t] += r1[t + st]; }
    __syncthreads();
  }
  if (t == 0) {
    float mu = r0[0] / HW;
    float var = (r1[0] - HW * mu * mu) / (HW - 1);
    mean[bc] = mu;
    sd[bc] = sqrtf(fmaxf(var, 0.f));
  }
}
__global__ __launch_bounds__(256) void k_mse_acc(const float* __restrict__ a,
                                                 const float* __restrict__ b,
                                                 float* __restrict__ out, long long n, float inv)
{
  __shared__ float red[256];
  long long i = (long long)blockIdx.x * 256 + threadIdx.x;
  long long st = (long long)gridDim.x * 256;
  float s = 0.f;
  for (; i < n; i += st) { float d = a[i] - b[i]; s += d * d; }
  red[threadIdx.x] = s; __syncthreads();
  for (int k = 128; k > 0; k >>= 1) { if (threadIdx.x < k) red[threadIdx.x] += red[threadIdx.x + k]; __syncthreads(); }
  if (threadIdx.x == 0) atomicAdd(out, red[0] * inv);
}

// ---------------------------------------------------------------------------
// host-side orchestration
// ---------------------------------------------------------------------------
static inline int cdiv(int a, int b) { return (a + b - 1) / b; }

static void conv3(const float* in, const float* w, const float* b, float* out,
                  int Cin, int Cout, int H, int W, int relu, hipStream_t s)
{
  dim3 g(H * W / 64, cdiv(Cout, 64), BATCH);
  size_t sm = 64 * 32 * 4 + 64 * 32 * 2;   // A f32 tile + B bf16 tile
  k_conv3x3<<<g, 256, sm, s>>>(in, w, b, out, Cin, Cout, H, W, relu);
}
static void gemm(const float* A, const float* B, float* C, int M, int N, int K,
                 int tB, const float* sp, float sc, float de, hipStream_t s)
{
  dim3 g(cdiv(N, 64), cdiv(M, 64));
  k_gemm_f32<<<g, 256, 0, s>>>(A, B, C, M, N, K, tB, sp, sc, de);
}
static void pool(const float* in, float* out, int C, int H, int W, hipStream_t s)
{
  int total = BATCH * C * (H / 2) * (W / 2);
  k_pool2<<<cdiv(total, 256), 256, 0, s>>>(in, out, C, H, W);
}
static void up(const float* in, float* out, int C, int H, int W, hipStream_t s)
{
  int total = BATCH * C * 2 * H * 2 * W;
  k_up2<<<cdiv(total, 256), 256, 0, s>>>(in, out, C, H, W);
}

// VGG encode; saves only h4; optional per-level mean/std stats.
static void encode_net(const float* img, const float* const* ew, const float* const* eb,
                       float* tA, float* tB, float* h4,
                       float* const* sm, float* const* sd, hipStream_t s)
{
  int tot0 = BATCH * 3 * 65536;
  k_conv1x1<<<cdiv(tot0, 256), 256, 0, s>>>(img, ew[0], eb[0], tA, 65536);
  conv3(tA, ew[1], eb[1], tB, 3, 64, 256, 256, 1, s);                      // h1 = tB
  if (sm) k_meanstd<<<BATCH * 64, 256, 0, s>>>(tB, sm[0], sd[0], 64, 65536);
  conv3(tB, ew[2], eb[2], tA, 64, 64, 256, 256, 1, s);
  pool(tA, tB, 64, 256, 256, s);
  conv3(tB, ew[3], eb[3], tA, 64, 128, 128, 128, 1, s);                    // h2 = tA
  if (sm) k_meanstd<<<BATCH * 128, 256, 0, s>>>(tA, sm[1], sd[1], 128, 16384);
  conv3(tA, ew[4], eb[4], tB, 128, 128, 128, 128, 1, s);
  pool(tB, tA, 128, 128, 128, s);
  conv3(tA, ew[5], eb[5], tB, 128, 256, 64, 64, 1, s);                     // h3 = tB
  if (sm) k_meanstd<<<BATCH * 256, 256, 0, s>>>(tB, sm[2], sd[2], 256, 4096);
  conv3(tB, ew[6], eb[6], tA, 256, 256, 64, 64, 1, s);
  conv3(tA, ew[7], eb[7], tB, 256, 256, 64, 64, 1, s);
  conv3(tB, ew[8], eb[8], tA, 256, 256, 64, 64, 1, s);
  pool(tA, tB, 256, 64, 64, s);
  conv3(tB, ew[9], eb[9], h4, 256, 512, 32, 32, 1, s);                     // h4
  if (sm) k_meanstd<<<BATCH * 512, 256, 0, s>>>(h4, sm[3], sd[3], 512, 1024);
}

static void decode_net(const float* cs, const float* const* dw, const float* const* db,
                       float* tA, float* tB, float* img, hipStream_t s)
{
  conv3(cs, dw[0], db[0], tA, 512, 256, 32, 32, 1, s);
  up(tA, tB, 256, 32, 32, s);
  conv3(tB, dw[1], db[1], tA, 256, 256, 64, 64, 1, s);
  conv3(tA, dw[2], db[2], tB, 256, 256, 64, 64, 1, s);
  conv3(tB, dw[3], db[3], tA, 256, 256, 64, 64, 1, s);
  conv3(tA, dw[4], db[4], tB, 256, 128, 64, 64, 1, s);
  up(tB, tA, 128, 64, 64, s);
  conv3(tA, dw[5], db[5], tB, 128, 128, 128, 128, 1, s);
  conv3(tB, dw[6], db[6], tA, 128, 64, 128, 128, 1, s);
  up(tA, tB, 64, 128, 128, s);
  conv3(tB, dw[7], db[7], tA, 64, 64, 256, 256, 1, s);
  conv3(tA, dw[8], db[8], img, 64, 3, 256, 256, 0, s);
}

extern "C" void kernel_launch(void* const* d_in, const int* in_sizes, int n_in,
                              void* d_out, int out_size, void* d_ws, size_t ws_size,
                              hipStream_t stream)
{
  (void)in_sizes; (void)n_in; (void)out_size; (void)ws_size;
  const float* content = (const float*)d_in[0];
  const float* style   = (const float*)d_in[1];
  const int*   clab    = (const int*)d_in[2];
  const int*   slab    = (const int*)d_in[3];
  const float* ew[10]; const float* eb[10];
  for (int i = 0; i < 10; ++i) { ew[i] = (const float*)d_in[4 + 2 * i]; eb[i] = (const float*)d_in[5 + 2 * i]; }
  const float* dw[9]; const float* db[9];
  for (int i = 0; i < 9; ++i)  { dw[i] = (const float*)d_in[24 + 2 * i]; db[i] = (const float*)d_in[25 + 2 * i]; }

  // workspace bump allocator (256-byte aligned)
  size_t off = 0;
  char* base = (char*)d_ws;
  auto alloc = [&](size_t nfl) -> float* {
    float* p = (float*)(base + off);
    off += ((nfl * sizeof(float) + 255) / 256) * 256;
    return p;
  };
  float* tA  = alloc((size_t)BATCH * 64 * 65536);   // ping
  float* tB  = alloc((size_t)BATCH * 64 * 65536);   // pong
  float* c4  = alloc((size_t)BATCH * 512 * 1024);
  float* s4  = alloc((size_t)BATCH * 512 * 1024);
  float* o4  = alloc((size_t)BATCH * 512 * 1024);
  float* cs  = alloc((size_t)BATCH * 512 * 1024);
  float* img = alloc((size_t)BATCH * 3 * 65536);
  float* Xc  = alloc(512 * 1024);
  float* Xs  = alloc(512 * 1024);
  float* cov = alloc(512 * 512);
  float* V   = alloc(512 * 512);
  float* VD  = alloc(512 * 512);
  float* VT  = alloc(512 * 512);
  float* T1  = alloc(512 * 1024);
  float* T2  = alloc(512 * 1024);
  float* WHT = alloc(512 * 1024);
  float* smean = alloc(512);
  float* cmean = alloc(512);
  float* invC  = alloc(1);
  float* invS  = alloc(1);
  const int CL[4] = {64, 128, 256, 512};
  float *sm[4], *sd[4], *om[4], *od[4];
  for (int l = 0; l < 4; ++l) { sm[l] = alloc(BATCH * CL[l]); sd[l] = alloc(BATCH * CL[l]);
                                om[l] = alloc(BATCH * CL[l]); od[l] = alloc(BATCH * CL[l]); }

  float* loss = (float*)d_out;
  k_zero<<<1, 64, 0, stream>>>(loss, 1);

  // --- encodes ---
  encode_net(content, ew, eb, tA, tB, c4, nullptr, nullptr, stream);       // c4
  encode_net(style,   ew, eb, tA, tB, s4, sm, sd, stream);                 // s4 + style stats

  // --- labeled WCT on relu4_1 ---
  k_zero<<<cdiv(BATCH * 512 * 1024, 256), 256, 0, stream>>>(cs, (long long)BATCH * 512 * 1024);
  for (int b = 0; b < BATCH; ++b) {
    for (int k = 0; k < 4; ++k) {
      k_masked_center<<<512, 256, 0, stream>>>(c4, clab, Xc, cmean, invC, b, k, 512, 1024);
      k_masked_center<<<512, 256, 0, stream>>>(s4, slab, Xs, smean, invS, b, k, 512, 1024);
      // content covariance -> whitening operator
      gemm(Xc, Xc, cov, 512, 512, 1024, 1, invC, 1.f, EPSV, stream);
      k_identity<<<cdiv(512 * 512, 256), 256, 0, stream>>>(V, 512);
      for (int sw = 0; sw < NSWEEP; ++sw) k_jacobi_sweep<<<1, 256, 0, stream>>>(cov, V, 512);
      k_eig_post<<<cdiv(512 * 512, 256), 256, 0, stream>>>(cov, V, VD, VT, 512, -0.5f);
      gemm(VT, Xc, T1, 512, 1024, 512, 0, nullptr, 1.f, 0.f, stream);       // V^T Xc
      gemm(VD, T1, WHT, 512, 1024, 512, 0, nullptr, 1.f, 0.f, stream);      // whitened
      // style covariance -> coloring operator
      gemm(Xs, Xs, cov, 512, 512, 1024, 1, invS, 1.f, EPSV, stream);
      k_identity<<<cdiv(512 * 512, 256), 256, 0, stream>>>(V, 512);
      for (int sw = 0; sw < NSWEEP; ++sw) k_jacobi_sweep<<<1, 256, 0, stream>>>(cov, V, 512);
      k_eig_post<<<cdiv(512 * 512, 256), 256, 0, stream>>>(cov, V, VD, VT, 512, 0.5f);
      gemm(VT, WHT, T1, 512, 1024, 512, 0, nullptr, 1.f, 0.f, stream);
      gemm(VD, T1, T2, 512, 1024, 512, 0, nullptr, 1.f, 0.f, stream);       // colored
      k_wct_accum<<<cdiv(512 * 1024, 256), 256, 0, stream>>>(
          cs + (size_t)b * 512 * 1024, T2, smean, clab, b, k, 512, 1024);
    }
  }

  // --- decode + re-encode + losses ---
  decode_net(cs, dw, db, tA, tB, img, stream);
  encode_net(img, ew, eb, tA, tB, o4, om, od, stream);                     // o4 + out stats

  long long nC = (long long)BATCH * 512 * 1024;
  k_mse_acc<<<512, 256, 0, stream>>>(o4, c4, loss, nC, 1.f / (float)nC);   // content loss
  for (int l = 0; l < 4; ++l) {                                            // style losses
    long long n = (long long)BATCH * CL[l];
    int g = cdiv((int)n, 256);
    k_mse_acc<<<g, 256, 0, stream>>>(om[l], sm[l], loss, n, 1.f / (float)n);
    k_mse_acc<<<g, 256, 0, stream>>>(od[l], sd[l], loss, n, 1.f / (float)n);
  }
}
